// RWKVBlock_78065325572196
// MI455X (gfx1250) — compile-verified, matrix operand order checked by reading, not run on hardware
//
#include <hip/hip_runtime.h>
#include <hip/hip_bf16.h>
#include <stdint.h>

// ---------------------------------------------------------------------------
// RWKV block for MI455X (gfx1250, wave32, WMMA + Tensor Data Mover).
// GEMMs: bf16 operands, f32 accumulation via v_wmma_f32_16x16x32_bf16.
// Operand staging: TDM tensor_load_to_lds with LDS padding (double-buffered).
// ---------------------------------------------------------------------------

typedef __attribute__((ext_vector_type(16))) __bf16 v16bf;
typedef __attribute__((ext_vector_type(8)))  float  v8f;
typedef __attribute__((ext_vector_type(4)))  int    v4i;
typedef __attribute__((ext_vector_type(8)))  int    v8i;
typedef __attribute__((ext_vector_type(4)))  unsigned int v4u;

#ifndef __has_builtin
#define __has_builtin(x) 0
#endif
#if __has_builtin(__builtin_amdgcn_tensor_load_to_lds) && \
    __has_builtin(__builtin_amdgcn_s_wait_tensorcnt)
#define USE_TDM 1
#else
#define USE_TDM 0
#endif

#define LN_THREADS   256
#define GEMM_THREADS 256
#define TM 128
#define TN 128
#define TK 32
#define LDST 40                    // halfs: 32 data + 8 pad = 80B rows (conflict-free b128)
#define TILE_BYTES (TM * LDST * 2) // 10240 B per tile buffer

__device__ __forceinline__ uint16_t f32_to_bf16(float f) {
    uint32_t u = __float_as_uint(f);
    u += 0x7FFFu + ((u >> 16) & 1u);     // round-to-nearest-even
    return (uint16_t)(u >> 16);
}

#if USE_TDM
// ---- TDM: async-load one 128-row x 64-byte 2D tile into LDS with padding --
// D# per CDNA5 ISA 8.3/8.4: data_size=2B, tile 32x128 elements, row stride K
// elements, pad 4 dwords after every 16 dwords (=> 80B LDS row stride).
__device__ __forceinline__ void tdm_load_tile(uint32_t lds_byte_off,
                                              const uint16_t* gptr,
                                              uint32_t k_elems) {
    const uint64_t ga = (uint64_t)(uintptr_t)gptr;
    v4u g0;
    g0.x = 1u;                                  // count=1 (valid user descriptor)
    g0.y = lds_byte_off;                        // lds_addr
    g0.z = (uint32_t)ga;                        // global_addr[31:0]
    g0.w = (uint32_t)((ga >> 32) & 0x01FFFFFFu) | (2u << 30);  // addr[56:32] | type=2

    v8i g1;
    g1[0] = (1 << 16)      // data_size = 2 bytes
          | (1 << 20)      // pad_enable
          | (3 << 22)      // pad_interval: every 16 dwords (64B)
          | (3 << 25);     // pad_amount: 4 dwords (16B)
    g1[1] = (int)(32u << 16);            // tensor_dim0 = 32 elems (bits 63:48)
    g1[2] = (int)(128u << 16);           // tensor_dim1 = 128 rows (bits 95:80)
    g1[3] = (int)(32u << 16);            // tile_dim0 = 32 elems (bits 127:112)
    g1[4] = (int)128u;                   // tile_dim1 = 128 rows; tile_dim2 = 0
    g1[5] = (int)k_elems;                // tensor_dim0_stride[31:0] (elems)
    g1[6] = 0;
    g1[7] = 0;

    v4i z4 = {0, 0, 0, 0};
#if defined(__clang_major__) && (__clang_major__ >= 23)
    v8i z8 = {0, 0, 0, 0, 0, 0, 0, 0};
    __builtin_amdgcn_tensor_load_to_lds(g0, g1, z4, z4, z8, 0);
#else
    __builtin_amdgcn_tensor_load_to_lds(g0, g1, z4, z4, 0);
#endif
}
#endif

// ---- fp32 -> bf16 conversion (weights / activations) ----------------------
__global__ __launch_bounds__(256)
void cvt_bf16_kernel(const float* __restrict__ in, uint16_t* __restrict__ out, int n) {
    int i = blockIdx.x * 256 + threadIdx.x;
    int stride = gridDim.x * 256;
    for (; i < n; i += stride) out[i] = f32_to_bf16(in[i]);
}

// ---- LayerNorm; optionally also emits time-shifted copy + final state -----
__global__ __launch_bounds__(LN_THREADS)
void ln_kernel(const float* __restrict__ x, const float* __restrict__ g,
               const float* __restrict__ beta,
               uint16_t* __restrict__ out_bf,     // (B,T,C) bf16
               uint16_t* __restrict__ shift_bf,   // (B,T,C) bf16 or nullptr
               float* __restrict__ state,         // (B,C) fp32 or nullptr
               int T, int C)
{
    const int t = blockIdx.x, b = blockIdx.y;
    const size_t rowoff = ((size_t)b * T + t) * (size_t)C;
    const float* row = x + rowoff;

    float s = 0.f, s2 = 0.f;
    for (int c = threadIdx.x; c < C; c += LN_THREADS) {
        float v = row[c]; s += v; s2 += v * v;
    }
    __shared__ float rs[LN_THREADS / 32], rs2[LN_THREADS / 32];
    __shared__ float smu, srstd;
    const int lane = threadIdx.x & 31, wid = threadIdx.x >> 5;
    for (int o = 16; o > 0; o >>= 1) {
        s  += __shfl_down(s,  o, 32);
        s2 += __shfl_down(s2, o, 32);
    }
    if (lane == 0) { rs[wid] = s; rs2[wid] = s2; }
    __syncthreads();
    if (threadIdx.x == 0) {
        float ts = 0.f, ts2 = 0.f;
        for (int i = 0; i < LN_THREADS / 32; ++i) { ts += rs[i]; ts2 += rs2[i]; }
        float mu  = ts / (float)C;
        float var = ts2 / (float)C - mu * mu;
        smu = mu;
        srstd = rsqrtf(var + 1e-5f);
    }
    __syncthreads();
    const float mu = smu, rstd = srstd;
    for (int c = threadIdx.x; c < C; c += LN_THREADS) {
        float y = (row[c] - mu) * rstd * g[c] + beta[c];
        uint16_t h = f32_to_bf16(y);
        out_bf[rowoff + c] = h;
        if (shift_bf) {
            if (t + 1 < T) shift_bf[rowoff + C + c] = h;   // shifted[b, t+1] = xln[b, t]
            if (t == 0)    shift_bf[(size_t)b * T * C + c] = 0;
            if (t == T - 2 && state) state[(size_t)b * C + c] = y;  // state = shifted[:, -1]
        }
    }
}

// ---- bf16 GEMM: out(M,N) = A(M,K) * W(N,K)^T, wave32 WMMA + TDM staging ---
// EPI 0: fp32 store.  EPI 1: bf16 store of relu(x)^2 (channel-mix k path).
template<int EPI>
__global__ __launch_bounds__(GEMM_THREADS)
void gemm_kernel(const uint16_t* __restrict__ A, const uint16_t* __restrict__ W,
                 void* __restrict__ outp, int M, int N, int K)
{
    __shared__ __align__(16) uint16_t lA[2][TM * LDST];
    __shared__ __align__(16) uint16_t lB[2][TN * LDST];

    const int tid  = threadIdx.x;
    const int wid  = tid >> 5;
    const int lane = tid & 31;

    const int tiles_n = N / TN;
    const int tile_m  = (int)(blockIdx.x / tiles_n) * TM;
    const int tile_n  = (int)(blockIdx.x % tiles_n) * TN;

    const int wm = (wid & 3) * 32;   // 4 waves along M (2 x 16-row tiles)
    const int wn = (wid >> 2) * 64;  // 2 waves along N (4 x 16-col tiles)

    v8f acc[2][4];
    #pragma unroll
    for (int i = 0; i < 2; ++i)
        #pragma unroll
        for (int j = 0; j < 4; ++j)
            acc[i][j] = (v8f){0.f,0.f,0.f,0.f,0.f,0.f,0.f,0.f};

    // 16-bit WMMA operand layout (ISA 7.12.2):
    //  lanes 0-15 hold K = [0..7] and [16..23], lanes 16-31 hold K = [8..15] and [24..31]
    const int lrow = lane & 15;
    const int klo  = (lane >> 4) << 3;   // 0 or 8 halfs

    union Frag { v16bf v; v4i q[2]; };

    const uint16_t* Abase = A + (size_t)tile_m * (size_t)K;
    const uint16_t* Wbase = W + (size_t)tile_n * (size_t)K;
    const int nk = K / TK;

#if USE_TDM
    const uint32_t ldsA0 = (uint32_t)(uintptr_t)(const void*)&lA[0][0];
    const uint32_t ldsB0 = (uint32_t)(uintptr_t)(const void*)&lB[0][0];
    // Prologue: DMA stage 0 into buffer 0 (wave 0 only; TENSORcnt tracks it).
    if (wid == 0) {
        tdm_load_tile(ldsA0, Abase, (uint32_t)K);
        tdm_load_tile(ldsB0, Wbase, (uint32_t)K);
    }
#endif

    for (int s = 0; s < nk; ++s) {
        const int buf = s & 1;
#if USE_TDM
        __syncthreads();   // everyone done reading buf^1 (previous stage)
        if (wid == 0) {
            if (s + 1 < nk) {
                // Prefetch next stage into the other buffer while we compute.
                tdm_load_tile(ldsA0 + (uint32_t)((buf ^ 1) * TILE_BYTES),
                              Abase + (size_t)(s + 1) * TK, (uint32_t)K);
                tdm_load_tile(ldsB0 + (uint32_t)((buf ^ 1) * TILE_BYTES),
                              Wbase + (size_t)(s + 1) * TK, (uint32_t)K);
                __builtin_amdgcn_s_wait_tensorcnt(2);   // stage s complete
            } else {
                __builtin_amdgcn_s_wait_tensorcnt(0);
            }
        }
        __syncthreads();   // publish stage s tiles to all waves
#else
        const int k0 = s * TK;
        __syncthreads();
        #pragma unroll
        for (int c = 0; c < 2; ++c) {
            int idx = tid + c * 256;            // 0..511
            int row = idx >> 2;
            int col = (idx & 3) << 3;           // halfs
            *(v4i*)&lA[buf][row * LDST + col] =
                *(const v4i*)(Abase + (size_t)row * (size_t)K + k0 + col);
            *(v4i*)&lB[buf][row * LDST + col] =
                *(const v4i*)(Wbase + (size_t)row * (size_t)K + k0 + col);
        }
        __syncthreads();
#endif

        Frag fa[2], fb[4];
        #pragma unroll
        for (int i = 0; i < 2; ++i) {
            const uint16_t* p = &lA[buf][(wm + i * 16 + lrow) * LDST + klo];
            fa[i].q[0] = *(const v4i*)(p);
            fa[i].q[1] = *(const v4i*)(p + 16);
        }
        #pragma unroll
        for (int j = 0; j < 4; ++j) {
            const uint16_t* p = &lB[buf][(wn + j * 16 + lrow) * LDST + klo];
            fb[j].q[0] = *(const v4i*)(p);
            fb[j].q[1] = *(const v4i*)(p + 16);
        }
        #pragma unroll
        for (int i = 0; i < 2; ++i)
            #pragma unroll
            for (int j = 0; j < 4; ++j)
                acc[i][j] = __builtin_amdgcn_wmma_f32_16x16x32_bf16(
                    false, fa[i].v, false, fb[j].v, (short)0, acc[i][j], false, false);
    }

    // Epilogue. C/D layout: lanes 0-15 -> M = e, N = lane; lanes 16-31 -> M = e+8.
    const int rbase = (lane >> 4) << 3;
    #pragma unroll
    for (int i = 0; i < 2; ++i)
        #pragma unroll
        for (int j = 0; j < 4; ++j)
            #pragma unroll
            for (int e = 0; e < 8; ++e) {
                const int grow = tile_m + wm + i * 16 + rbase + e;
                const int gcol = tile_n + wn + j * 16 + lrow;
                const float val = acc[i][j][e];
                if (EPI == 0) {
                    ((float*)outp)[(size_t)grow * (size_t)N + gcol] = val;
                } else {
                    float t = val > 0.f ? val : 0.f;
                    ((uint16_t*)outp)[(size_t)grow * (size_t)N + gcol] = f32_to_bf16(t * t);
                }
            }
}

// ---- WKV recurrence: serial over T, parallel over B*C channels ------------
__global__ __launch_bounds__(256)
void wkv_kernel(const float* __restrict__ k, const float* __restrict__ v,
                const float* __restrict__ r, const float* __restrict__ x,
                const float* __restrict__ time_decay,
                float* __restrict__ x_mid, int T, int C)
{
    const int gid = blockIdx.x * 256 + threadIdx.x;   // 0 .. B*C-1
    const int b = gid / C, c = gid % C;
    const float decay = __expf(-__expf(time_decay[c]));
    float a = 0.f, bb = 0.f;
    size_t idx = (size_t)b * T * C + c;
    for (int t = 0; t < T; ++t, idx += C) {
        const float ek = __expf(k[idx]);
        a  = decay * a  + ek * v[idx];
        bb = decay * bb + ek;
        const float wkv = a / (bb + 1e-8f);
        x_mid[idx] = x[idx] + r[idx] * wkv;
    }
}

// ---- final combine: out = x_mid + sigmoid(cr) * dv ------------------------
__global__ __launch_bounds__(256)
void combine_kernel(const float* __restrict__ xm, const float* __restrict__ cr,
                    const float* __restrict__ dv, float* __restrict__ out, int n)
{
    const int i = blockIdx.x * 256 + threadIdx.x;
    if (i < n) {
        const float s = 1.f / (1.f + __expf(-cr[i]));
        out[i] = xm[i] + s * dv[i];
    }
}

// ---------------------------------------------------------------------------
extern "C" void kernel_launch(void* const* d_in, const int* in_sizes, int n_in,
                              void* d_out, int out_size, void* d_ws, size_t ws_size,
                              hipStream_t stream)
{
    (void)in_sizes; (void)n_in; (void)out_size; (void)ws_size;
    const int Bn = 4, T = 2048, C = 2048, FF = 8192;
    const int M = Bn * T;   // 8192 rows

    const float* x   = (const float*)d_in[0];
    const float* g1  = (const float*)d_in[1];
    const float* b1  = (const float*)d_in[2];
    const float* td  = (const float*)d_in[3];
    const float* Wk  = (const float*)d_in[4];
    const float* Wv  = (const float*)d_in[5];
    const float* Wr  = (const float*)d_in[6];
    const float* g2  = (const float*)d_in[7];
    const float* b2  = (const float*)d_in[8];
    const float* Wck = (const float*)d_in[9];
    const float* Wcv = (const float*)d_in[10];
    const float* Wcr = (const float*)d_in[11];

    float* x_out = (float*)d_out;
    float* state = x_out + (size_t)M * C;

    char* ws = (char*)d_ws;
    size_t off = 0;
    auto take = [&](size_t bytes) -> char* {
        char* p = ws + off;
        off += (bytes + 255) & ~(size_t)255;
        return p;
    };

    uint16_t* xln  = (uint16_t*)take((size_t)M * C * 2);   // later reused for x2
    uint16_t* xlns = (uint16_t*)take((size_t)M * C * 2);
    uint16_t* ckb  = (uint16_t*)take((size_t)M * FF * 2);
    float*    kf   = (float*)take((size_t)M * C * 4);      // later reused for dv
    float*    vf   = (float*)take((size_t)M * C * 4);      // later reused for cr
    float*    rf   = (float*)take((size_t)M * C * 4);
    float*    xmid = (float*)take((size_t)M * C * 4);
    uint16_t* Wk_h  = (uint16_t*)take((size_t)C * C * 2);
    uint16_t* Wv_h  = (uint16_t*)take((size_t)C * C * 2);
    uint16_t* Wr_h  = (uint16_t*)take((size_t)C * C * 2);
    uint16_t* Wcr_h = (uint16_t*)take((size_t)C * C * 2);
    uint16_t* Wck_h = (uint16_t*)take((size_t)FF * C * 2);
    uint16_t* Wcv_h = (uint16_t*)take((size_t)C * FF * 2);

    auto cvt = [&](const float* src, uint16_t* dst, size_t n) {
        int blocks = (int)((n + 1023) / 1024);
        cvt_bf16_kernel<<<blocks, 256, 0, stream>>>(src, dst, (int)n);
    };
    auto gemm = [&](const uint16_t* A, const uint16_t* W, void* out,
                    int m, int n, int k, int epi) {
        dim3 grid((unsigned)((m / TM) * (n / TN)));
        if (epi == 0)
            gemm_kernel<0><<<grid, GEMM_THREADS, 0, stream>>>(A, W, out, m, n, k);
        else
            gemm_kernel<1><<<grid, GEMM_THREADS, 0, stream>>>(A, W, out, m, n, k);
    };

    // 1. bf16 weight conversion
    cvt(Wk,  Wk_h,  (size_t)C * C);
    cvt(Wv,  Wv_h,  (size_t)C * C);
    cvt(Wr,  Wr_h,  (size_t)C * C);
    cvt(Wcr, Wcr_h, (size_t)C * C);
    cvt(Wck, Wck_h, (size_t)FF * C);
    cvt(Wcv, Wcv_h, (size_t)C * FF);

    // 2. LN1 -> xln (bf16), shifted copy, state
    {
        dim3 grid((unsigned)T, (unsigned)Bn);
        ln_kernel<<<grid, LN_THREADS, 0, stream>>>(x, g1, b1, xln, xlns, state, T, C);
    }

    // 3. k / v (shifted input) and r (unshifted) projections
    gemm(xlns, Wk_h, kf, M, C, C, 0);
    gemm(xlns, Wv_h, vf, M, C, C, 0);
    gemm(xln,  Wr_h, rf, M, C, C, 0);

    // 4. WKV scan + residual -> x_mid
    wkv_kernel<<<(Bn * C) / 256, 256, 0, stream>>>(kf, vf, rf, x, td, xmid, T, C);

    // 5. LN2 -> x2 (bf16, reusing xln buffer)
    {
        dim3 grid((unsigned)T, (unsigned)Bn);
        ln_kernel<<<grid, LN_THREADS, 0, stream>>>(xmid, g2, b2, xln, nullptr, nullptr, T, C);
    }

    // 6. channel mix: ck = relu(x2 @ Wck^T)^2 (bf16), dv = ck @ Wcv^T, cr = x2 @ Wcr^T
    gemm(xln, Wck_h, ckb, M, FF, C, 1);
    gemm(ckb, Wcv_h, kf,  M, C, FF, 0);   // dv reuses kf
    gemm(xln, Wcr_h, vf,  M, C, C, 0);    // cr (pre-sigmoid) reuses vf

    // 7. out = x_mid + sigmoid(cr) * dv
    {
        const int n = M * C;
        combine_kernel<<<(n + 255) / 256, 256, 0, stream>>>(xmid, vf, kf, x_out, n);
    }
}
